// FCM_64836826300506
// MI455X (gfx1250) — matrix operand order repaired and checked by reading, not compiled, not run on hardware
//
#include <hip/hip_runtime.h>
#include <math.h>

// ---------------------------------------------------------------------------
// FCM: scores = F F^T ; top-16/row ; softmax over top-16 ; att = W_sparse @ F ;
// out = l2norm(F*w + att).  N=8192, D=512, k=16.
// bf16 WMMA (16x16x32). Each wave computes 4 column tiles per pass so one
// LDS A-fragment load feeds 4 WMMAs (4 independent accumulator chains).
// Low register pressure lets the scheduler keep many B loads in flight
// (round-1 style partial waits) instead of draining loadcnt per WMMA.
// ---------------------------------------------------------------------------

typedef __attribute__((ext_vector_type(16))) __bf16        v16bf;
typedef __attribute__((ext_vector_type(8)))  float         v8f;
typedef __attribute__((ext_vector_type(4)))  unsigned int  u32x4;

#define NROWS   8192
#define DIM     512
#define TOPK    16
#define JTILES  (NROWS / 16)   // 512 column tiles
#define KCHUNKS (DIM / 32)     // 16 k-chunks of 32
#define WAVES   8              // 256 threads / wave32

__device__ __forceinline__ unsigned short f32_to_bf16(float f) {
    unsigned int u = __float_as_uint(f);
    unsigned int r = (u + 0x7FFFu + ((u >> 16) & 1u)) >> 16;  // RNE
    return (unsigned short)r;
}

__global__ void cvt_bf16(const float* __restrict__ in,
                         unsigned short* __restrict__ out, int n) {
    int i = blockIdx.x * blockDim.x + threadIdx.x;
    if (i < n) out[i] = f32_to_bf16(in[i]);
}

// LDS layout (bytes):
//   [0     ..16384) Abf   : strip A, 16x512 bf16          (phase 1)
//   [16384 ..32768) sc    : per-wave 2x(16x16) f32 bounce (phase 1, used twice/pass)
//   [0     ..32768) fcm   : 16x512 f32                    (phase 2, aliases above)
//   [32768 ..40960) wv    : per-wave top-16 values [8][16][16] f32
//   [40960 ..49152) wi    : per-wave top-16 indices
//   [49152 ..50176) fwgt  : softmax weights [16][16]
//   [50176 ..51200) fidx  : final indices   [16][16]
//   [51200 ..51264) nrm   : row sum-of-squares [16]
#define SMEM_BYTES 51264

__global__ __launch_bounds__(256)
void fcm_topk_attn(
    const float* __restrict__ feats,          // [8192][512] f32
    const unsigned short* __restrict__ fbf,   // [8192][512] bf16
    const int* __restrict__ wscale,           // scalar w
    float* __restrict__ out)                  // [8192][512] f32
{
    __shared__ __align__(16) unsigned char smem[SMEM_BYTES];
    unsigned short* Abf = (unsigned short*)(smem);
    float*          sc  = (float*)(smem + 16384);
    float*          fcm = (float*)(smem);
    float*          wv  = (float*)(smem + 32768);
    int*            wi  = (int*)  (smem + 40960);
    float*          fwgt= (float*)(smem + 49152);
    int*            fidx= (int*)  (smem + 50176);
    float*          nrm = (float*)(smem + 51200);

    const int tid   = threadIdx.x;
    const int wave  = tid >> 5;
    const int lane  = tid & 31;
    const int kh    = lane >> 4;      // lane half (0/1)
    const int mr    = lane & 15;
    const int strip = blockIdx.x;
    const int row0  = strip * 16;

    if (tid < 16) nrm[tid] = 0.0f;
    if (lane < 16) {
        float* vals = wv + (wave * 16 + lane) * 16;
        int*   idxs = wi + (wave * 16 + lane) * 16;
#pragma unroll
        for (int t = 0; t < TOPK; ++t) { vals[t] = -3.402823466e38f; idxs[t] = 0; }
    }

    // Stage the 16-row A strip into LDS as bf16.
    for (int i = tid; i < 16 * DIM; i += 256)
        Abf[i] = f32_to_bf16(feats[(size_t)row0 * DIM + i]);
    __syncthreads();

    union Frag { v16bf v; u32x4 u[2]; };

    // ---- Phase 1: stream column tiles (4 per pass), keep per-wave top-16 ----
    float* scw = sc + wave * 512;
    const unsigned short* Arow = Abf + mr * DIM + kh * 16;

    for (int q = 0; q < JTILES / (4 * WAVES); ++q) {   // 16 passes
        const int jt0 = wave * 4 + q * 32;             // tiles jt0 .. jt0+3
        v8f c0 = {0.f, 0.f, 0.f, 0.f, 0.f, 0.f, 0.f, 0.f};
        v8f c1 = {0.f, 0.f, 0.f, 0.f, 0.f, 0.f, 0.f, 0.f};
        v8f c2 = {0.f, 0.f, 0.f, 0.f, 0.f, 0.f, 0.f, 0.f};
        v8f c3 = {0.f, 0.f, 0.f, 0.f, 0.f, 0.f, 0.f, 0.f};
        // B 32x16 bf16 layout: lanes 0-15 col=mr K=kb+0..15,
        //                      lanes 16-31 col=mr K=kb+16..31
        const unsigned short* Brow0 = fbf + (size_t)(jt0 * 16 + mr) * DIM;
        const unsigned short* Brow1 = Brow0 + 16 * DIM;
        const unsigned short* Brow2 = Brow0 + 32 * DIM;
        const unsigned short* Brow3 = Brow0 + 48 * DIM;

#pragma unroll
        for (int kc = 0; kc < KCHUNKS; ++kc) {
            const int kb = kc * 32;
            // A 16x32 bf16 layout: lane-half 0 -> K kb+0..15,
            //                      lane-half 1 -> K kb+16..31
            Frag a;
            a.u[0] = *(const u32x4*)(Arow + kb);
            a.u[1] = *(const u32x4*)(Arow + kb + 8);
            const int kbb = kb + kh * 16;
            Frag b0, b1, b2, b3;
            b0.u[0] = *(const u32x4*)(Brow0 + kbb);
            b0.u[1] = *(const u32x4*)(Brow0 + kbb + 8);
            b1.u[0] = *(const u32x4*)(Brow1 + kbb);
            b1.u[1] = *(const u32x4*)(Brow1 + kbb + 8);
            b2.u[0] = *(const u32x4*)(Brow2 + kbb);
            b2.u[1] = *(const u32x4*)(Brow2 + kbb + 8);
            b3.u[0] = *(const u32x4*)(Brow3 + kbb);
            b3.u[1] = *(const u32x4*)(Brow3 + kbb + 8);
            c0 = __builtin_amdgcn_wmma_f32_16x16x32_bf16(
                     false, a.v, false, b0.v, (short)0, c0, false, false);
            c1 = __builtin_amdgcn_wmma_f32_16x16x32_bf16(
                     false, a.v, false, b1.v, (short)0, c1, false, false);
            c2 = __builtin_amdgcn_wmma_f32_16x16x32_bf16(
                     false, a.v, false, b2.v, (short)0, c2, false, false);
            c3 = __builtin_amdgcn_wmma_f32_16x16x32_bf16(
                     false, a.v, false, b3.v, (short)0, c3, false, false);
        }

        // Top-k: two batches of two tiles through the same LDS bounce.
#pragma unroll
        for (int pair = 0; pair < 2; ++pair) {
            // Scatter C (VGPR v, lane) -> row v+8*kh, col mr.
#pragma unroll
            for (int v = 0; v < 8; ++v) {
                scw[(v + 8 * kh) * 16 + mr]       = (pair == 0) ? c0[v] : c2[v];
                scw[256 + (v + 8 * kh) * 16 + mr] = (pair == 0) ? c1[v] : c3[v];
            }
            __builtin_amdgcn_wave_barrier();
            if (lane < 16) {
                float* vals = wv + (wave * 16 + lane) * 16;
                int*   idxs = wi + (wave * 16 + lane) * 16;
#pragma unroll
                for (int tile = 0; tile < 2; ++tile) {
                    const float* srow  = scw + tile * 256 + lane * 16;
                    const int    jbase = (jt0 + 2 * pair + tile) * 16;
                    for (int cc = 0; cc < 16; ++cc) {
                        float v = srow[cc];
                        if (v > vals[TOPK - 1]) {
                            int j = jbase + cc;
                            int p = TOPK - 1;
                            while (p > 0 && vals[p - 1] < v) {
                                vals[p] = vals[p - 1]; idxs[p] = idxs[p - 1]; --p;
                            }
                            vals[p] = v; idxs[p] = j;
                        }
                    }
                }
            }
            __builtin_amdgcn_wave_barrier();
        }
    }
    __syncthreads();

    // ---- Phase 1b: 8-way merge of sorted lists, softmax over top-16 ----
    if (tid < 16) {
        const int r = tid;
        int heads[WAVES] = {0, 0, 0, 0, 0, 0, 0, 0};
        float tv[TOPK];
        int   ti[TOPK];
#pragma unroll
        for (int t = 0; t < TOPK; ++t) {
            float best = -3.402823466e38f; int bw = 0;
#pragma unroll
            for (int w2 = 0; w2 < WAVES; ++w2) {
                int h = heads[w2];
                if (h < TOPK) {
                    float v = wv[(w2 * 16 + r) * 16 + h];
                    if (v > best) { best = v; bw = w2; }
                }
            }
            tv[t] = best;
            ti[t] = wi[(bw * 16 + r) * 16 + heads[bw]];
            heads[bw]++;
        }
        const float m = tv[0];          // sorted descending
        float s = 0.0f;
#pragma unroll
        for (int t = 0; t < TOPK; ++t) { tv[t] = __expf(tv[t] - m); s += tv[t]; }
        const float inv = 1.0f / s;
#pragma unroll
        for (int t = 0; t < TOPK; ++t) {
            fwgt[r * 16 + t] = tv[t] * inv;
            fidx[r * 16 + t] = ti[t];
        }
    }
    __syncthreads();

    // ---- Phase 2: att = sum_t w_t * feats[idx_t], residual, L2 normalize ----
    {
        const int r   = tid >> 4;   // 16 threads per row
        const int l16 = tid & 15;
        float wgts[TOPK]; int idxs[TOPK];
#pragma unroll
        for (int t = 0; t < TOPK; ++t) {
            wgts[t] = fwgt[r * 16 + t];
            idxs[t] = fidx[r * 16 + t];
        }
        const float  wf   = (float)wscale[0];
        const float* frow = feats + (size_t)(row0 + r) * DIM;
        float ss = 0.0f;
        for (int q = 0; q < DIM / 16; ++q) {
            const int d = l16 + q * 16;
            float acc = frow[d] * wf;
#pragma unroll
            for (int t = 0; t < TOPK; ++t)
                acc += wgts[t] * feats[(size_t)idxs[t] * DIM + d];
            fcm[r * DIM + d] = acc;
            ss += acc * acc;
        }
        atomicAdd(&nrm[r], ss);
        __syncthreads();
        const float inv = 1.0f / fmaxf(sqrtf(nrm[r]), 1e-12f);
        float* orow = out + (size_t)(row0 + r) * DIM;
        for (int q = 0; q < DIM / 16; ++q) {
            const int d = l16 + q * 16;
            orow[d] = fcm[r * DIM + d] * inv;
        }
    }
}

extern "C" void kernel_launch(void* const* d_in, const int* in_sizes, int n_in,
                              void* d_out, int out_size, void* d_ws, size_t ws_size,
                              hipStream_t stream) {
    (void)in_sizes; (void)n_in; (void)out_size; (void)ws_size;
    const float* feats = (const float*)d_in[0];
    // d_in[1] = k (fixed 16), d_in[2] = w
    const int*   wptr  = (const int*)d_in[2];
    float*       out   = (float*)d_out;
    unsigned short* fbf = (unsigned short*)d_ws;   // 8 MB bf16 copy of feats

    const int total = NROWS * DIM;
    cvt_bf16<<<(total + 255) / 256, 256, 0, stream>>>(feats, fbf, total);
    fcm_topk_attn<<<NROWS / 16, 256, 0, stream>>>(feats, fbf, wptr, out);
}
